// MatchLSTM_21122649162545
// MI455X (gfx1250) — matrix operand order repaired and checked by reading, not compiled, not run on hardware
//
#include <hip/hip_runtime.h>
#include <hip/hip_bf16.h>

typedef __attribute__((ext_vector_type(16))) _Float16 v16h;
typedef __attribute__((ext_vector_type(8)))  float    v8f;

namespace {
constexpr int kSP = 64;
constexpr int kSH = 32;
constexpr int kB  = 128;
constexpr int kE  = 300;
constexpr int kH  = 512;
constexpr int kG  = 2048;                 // 4*H
constexpr int KB_E = 10;                  // E padded to 320 -> 10 k-blocks of 32
constexpr int KB_H = 16;                  // 512/32
constexpr int FRAG = 512;                 // halves per 16x16(x32K) fragment (32 lanes * 16)
constexpr int STEP_FRAG_H = (kB / 16) * KB_H * FRAG;   // 65536 halves per [128,512] block
constexpr int STEP_FRAG_X = (kB / 16) * KB_E * FRAG;   // 40960 halves per [128,320] block
}

// ---------------------------------------------------------------------------
// Fragment layout helpers (per CDNA5 ISA 7.12.2, wave32)
// A operand (16-bit, 16x32): lane<16 holds K {0..7,16..23}; lane>=16 {8..15,24..31}
__device__ __forceinline__ int koff_a(int lane, int e) {
  return (((lane >> 4) & 1) << 3) + (e & 7) + (((e >> 3) & 1) << 4);
}
// B operand (16-bit, 32x16): lane<16 holds K {0..7,16..23}; lane>=16 {8..15,24..31}
__device__ __forceinline__ int koff_b(int lane, int e) {
  return (((lane >> 4) & 1) << 3) + (e & 7) + (((e >> 3) & 1) << 4);
}
// Scatter h[b, hidx] into an A-operand fragment buffer [(mt*KB_H+kb)*512 + lane*16 + e]
__device__ __forceinline__ void write_afrag_h(_Float16* f, int b, int h, _Float16 v) {
  int mt = b >> 4;
  int kb = h >> 5;
  int kp = h & 31;
  int lane = (b & 15) + (((kp >> 3) & 1) << 4);
  int e = (kp & 7) + (((kp >> 4) & 1) << 3);
  f[(size_t)(mt * KB_H + kb) * FRAG + lane * 16 + e] = v;
}

__device__ __forceinline__ float sigf(float x) { return 1.0f / (1.0f + __expf(-x)); }

// ---------------------------------------------------------------------------
// Pack W [Nw, Kw] (row-major f32) as B-operand fragments of W^T [K, N] into dst,
// occupying k-blocks [kb_off, kb_off+KBseg) of a combined-K fragment buffer.
__global__ __launch_bounds__(256) void pack_b_seg(const float* __restrict__ W, int Nw, int Kw,
                                                  _Float16* __restrict__ dst, int kb_off,
                                                  int KBtot, int KBseg, int Ntiles) {
  int idx = blockIdx.x * blockDim.x + threadIdx.x;
  int total = Ntiles * KBseg * 32;
  if (idx >= total) return;
  int lane = idx & 31;
  int rest = idx >> 5;
  int kb_l = rest % KBseg;
  int nt = rest / KBseg;
  int n = nt * 16 + (lane & 15);
  _Float16* o = dst + (size_t)(nt * KBtot + kb_off + kb_l) * FRAG + lane * 16;
#pragma unroll
  for (int e = 0; e < 16; ++e) {
    int k = kb_l * 32 + koff_b(lane, e);
    float v = (k < Kw && n < Nw) ? W[(size_t)n * Kw + k] : 0.0f;
    o[e] = (_Float16)v;
  }
}

// Gather embeddings into A-operand fragments: rows = T*B token rows, K padded to 320.
__global__ __launch_bounds__(256) void gather_embed(const int* __restrict__ tok, int rows,
                                                    const float* __restrict__ table,
                                                    _Float16* __restrict__ dst) {
  int idx = blockIdx.x * blockDim.x + threadIdx.x;
  int total = (rows / 16) * KB_E * 32;
  if (idx >= total) return;
  int lane = idx & 31;
  int rest = idx >> 5;
  int kb = rest % KB_E;
  int mt = rest / KB_E;
  int r = mt * 16 + (lane & 15);
  const float* emb = table + (size_t)tok[r] * kE;
  _Float16* o = dst + (size_t)(mt * KB_E + kb) * FRAG + lane * 16;
#pragma unroll
  for (int e = 0; e < 16; ++e) {
    int k = kb * 32 + koff_a(lane, e);
    o[e] = (k < kE) ? (_Float16)emb[k] : (_Float16)0.0f;
  }
}

// ---------------------------------------------------------------------------
// One compile-time K-segment of the fragment GEMM.  KBs/KBtot/KBbase are
// constants, so the fully-unrolled loop folds all B offsets into the
// instruction immediate offsets of the b128 loads.  All four B fragments are
// loaded into registers before the WMMA quartet so the scheduler can overlap
// outstanding loads with earlier WMMAs (partial s_wait_loadcnt) instead of
// draining loads before every WMMA.
template <int KBs, int KBtot, int KBbase>
__device__ __forceinline__ void gemm_seg(const _Float16* __restrict__ As, int mt, int lane,
                                         const _Float16* __restrict__ Bf, int nt0,
                                         v8f (&acc)[4]) {
  if constexpr (KBs > 0) {
    const _Float16* ap = As + (size_t)mt * (KBs * FRAG) + lane * 16;
    const _Float16* bp = Bf + (size_t)nt0 * (KBtot * FRAG) + lane * 16;
#pragma unroll
    for (int kl = 0; kl < KBs; ++kl) {
      v16h a = *(const v16h*)(ap + kl * FRAG);
      __builtin_prefetch(ap + (kl + 1) * FRAG, 0, 1);
      v16h b0 = *(const v16h*)(bp + (size_t)0 * (KBtot * FRAG) + (KBbase + kl) * FRAG);
      v16h b1 = *(const v16h*)(bp + (size_t)1 * (KBtot * FRAG) + (KBbase + kl) * FRAG);
      v16h b2 = *(const v16h*)(bp + (size_t)2 * (KBtot * FRAG) + (KBbase + kl) * FRAG);
      v16h b3 = *(const v16h*)(bp + (size_t)3 * (KBtot * FRAG) + (KBbase + kl) * FRAG);
      acc[0] = __builtin_amdgcn_wmma_f32_16x16x32_f16(false, a, false, b0, (short)0, acc[0],
                                                      false, false);
      acc[1] = __builtin_amdgcn_wmma_f32_16x16x32_f16(false, a, false, b1, (short)0, acc[1],
                                                      false, false);
      acc[2] = __builtin_amdgcn_wmma_f32_16x16x32_f16(false, a, false, b2, (short)0, acc[2],
                                                      false, false);
      acc[3] = __builtin_amdgcn_wmma_f32_16x16x32_f16(false, a, false, b3, (short)0, acc[3],
                                                      false, false);
    }
  }
}

// Generic fragment GEMM: C[M,N] = bias + (A0|A1|A2) x B.  All operands packed.
// Each wave computes one 16-row M-tile x 4 consecutive 16-col N-tiles,
// reusing the A fragment across the 4 WMMAs per k-block.
template <int KB0, int KB1, int KB2>
__global__ __launch_bounds__(256) void wmma_gemm(const _Float16* __restrict__ A0,
                                                 const _Float16* __restrict__ A1,
                                                 const _Float16* __restrict__ A2,
                                                 const _Float16* __restrict__ Bf,
                                                 const float* __restrict__ bias,
                                                 float* __restrict__ Cout,
                                                 int Mtiles, int Ntiles, int ldc) {
  constexpr int KBtot = KB0 + KB1 + KB2;
  const int lane = threadIdx.x & 31;
  const int wave = threadIdx.x >> 5;
  const int ngrp = Ntiles >> 2;
  const int g = blockIdx.x * (blockDim.x >> 5) + wave;
  if (g >= Mtiles * ngrp) return;
  const int mt = g / ngrp;
  const int nt0 = (g % ngrp) << 2;

  v8f acc[4];
#pragma unroll
  for (int j = 0; j < 4; ++j) {
    float bv = bias ? bias[(nt0 + j) * 16 + (lane & 15)] : 0.0f;
#pragma unroll
    for (int v = 0; v < 8; ++v) acc[j][v] = bv;
  }

  gemm_seg<KB0, KBtot, 0>(A0, mt, lane, Bf, nt0, acc);
  gemm_seg<KB1, KBtot, KB0>(A1, mt, lane, Bf, nt0, acc);
  gemm_seg<KB2, KBtot, KB0 + KB1>(A2, mt, lane, Bf, nt0, acc);

  // C/D layout: VGPR v -> row (mt*16 + v + 8*(lane>=16)), col = nt*16 + (lane&15)
  const int rowp = mt * 16 + ((lane >> 4) << 3);
  const int col = lane & 15;
#pragma unroll
  for (int j = 0; j < 4; ++j) {
    float* cp = Cout + (size_t)rowp * ldc + (nt0 + j) * 16 + col;
#pragma unroll
    for (int v = 0; v < 8; ++v) cp[(size_t)v * ldc] = acc[j][v];
  }
}

// ---------------------------------------------------------------------------
// LSTM elementwise update (gate order i,f,g,o).  Writes the carry-h directly in
// A-fragment layout for the next step's GEMM; optional masked copies for the
// stored encoder outputs (reference masks outputs but not the carry).
__global__ __launch_bounds__(256) void lstm_pointwise(const float* __restrict__ gates,
                                                      float* __restrict__ c_state,
                                                      _Float16* __restrict__ h_carry_frag,
                                                      int t, const int* __restrict__ len,
                                                      float* __restrict__ h_masked_f32,
                                                      _Float16* __restrict__ h_masked_frag) {
  int idx = blockIdx.x * blockDim.x + threadIdx.x;
  if (idx >= kB * kH) return;
  int b = idx >> 9;
  int h = idx & (kH - 1);
  const float* gb = gates + (size_t)b * kG;
  float gi = gb[h], gf = gb[kH + h], gg = gb[2 * kH + h], go = gb[3 * kH + h];
  float c = c_state[idx];
  float cn = sigf(gf) * c + sigf(gi) * tanhf(gg);
  float hn = sigf(go) * tanhf(cn);
  c_state[idx] = cn;
  write_afrag_h(h_carry_frag, b, h, (_Float16)hn);
  float hm = (t < len[b]) ? hn : 0.0f;
  if (h_masked_f32) h_masked_f32[idx] = hm;
  if (h_masked_frag) write_afrag_h(h_masked_frag, b, h, (_Float16)hm);
}

__global__ __launch_bounds__(256) void match_pointwise(const float* __restrict__ gates,
                                                       float* __restrict__ c_state,
                                                       _Float16* __restrict__ hm_frag,
                                                       float* __restrict__ h_last,
                                                       int k, const int* __restrict__ len) {
  int idx = blockIdx.x * blockDim.x + threadIdx.x;
  if (idx >= kB * kH) return;
  int b = idx >> 9;
  int h = idx & (kH - 1);
  const float* gb = gates + (size_t)b * kG;
  float gi = gb[h], gf = gb[kH + h], gg = gb[2 * kH + h], go = gb[3 * kH + h];
  float c = c_state[idx];
  float cn = sigf(gf) * c + sigf(gi) * tanhf(gg);
  float hn = sigf(go) * tanhf(cn);
  c_state[idx] = cn;
  write_afrag_h(hm_frag, b, h, (_Float16)hn);
  if (k + 1 == len[b]) h_last[idx] = hn;
}

// ---------------------------------------------------------------------------
// Attention: e[s,b] = sum_h w_e[h] * tanh(WsHs[s,b,h] + q[b,h])  (one wave per (s,b))
__global__ __launch_bounds__(256) void attn_e(const float* __restrict__ WsHs,
                                              const float* __restrict__ q,
                                              const float* __restrict__ w_e,
                                              float* __restrict__ e_out) {
  int wid = (blockIdx.x * blockDim.x + threadIdx.x) >> 5;
  int lane = threadIdx.x & 31;
  if (wid >= kSP * kB) return;
  int s = wid >> 7;
  int b = wid & (kB - 1);
  const float* ws = WsHs + ((size_t)s * kB + b) * kH;
  const float* qb = q + (size_t)b * kH;
  float sum = 0.0f;
  for (int h = lane; h < kH; h += 32) sum += w_e[h] * tanhf(ws[h] + qb[h]);
#pragma unroll
  for (int off = 16; off > 0; off >>= 1) sum += __shfl_xor(sum, off, 32);
  if (lane == 0) e_out[s * kB + b] = sum;
}

// Softmax over s (64 premise positions); one wave per batch element.
__global__ __launch_bounds__(256) void attn_softmax(const float* __restrict__ e,
                                                    float* __restrict__ alpha) {
  int wid = (blockIdx.x * blockDim.x + threadIdx.x) >> 5;
  int lane = threadIdx.x & 31;
  if (wid >= kB) return;
  int b = wid;
  float v0 = e[lane * kB + b];
  float v1 = e[(lane + 32) * kB + b];
  float mx = fmaxf(v0, v1);
#pragma unroll
  for (int off = 16; off > 0; off >>= 1) mx = fmaxf(mx, __shfl_xor(mx, off, 32));
  float e0 = __expf(v0 - mx), e1 = __expf(v1 - mx);
  float sm = e0 + e1;
#pragma unroll
  for (int off = 16; off > 0; off >>= 1) sm += __shfl_xor(sm, off, 32);
  float inv = 1.0f / sm;
  alpha[lane * kB + b] = e0 * inv;
  alpha[(lane + 32) * kB + b] = e1 * inv;
}

// a_k[b,h] = sum_s alpha[s,b] * h_s[s,b,h]; written straight into A-fragment form.
__global__ __launch_bounds__(256) void attn_ak(const float* __restrict__ alpha,
                                               const float* __restrict__ hs,
                                               _Float16* __restrict__ a_frag) {
  int idx = blockIdx.x * blockDim.x + threadIdx.x;
  if (idx >= kB * kH) return;
  int b = idx >> 9;
  int h = idx & (kH - 1);
  float s = 0.0f;
  for (int sp = 0; sp < kSP; ++sp)
    s += alpha[sp * kB + b] * hs[((size_t)sp * kB + b) * kH + h];
  write_afrag_h(a_frag, b, h, (_Float16)s);
}

__global__ __launch_bounds__(256) void fc_out(const float* __restrict__ h_last,
                                              const float* __restrict__ W,
                                              const float* __restrict__ bias,
                                              float* __restrict__ out) {
  int idx = blockIdx.x * blockDim.x + threadIdx.x;
  if (idx >= kB * 3) return;
  int b = idx / 3, c = idx % 3;
  float s = bias[c];
  for (int h = 0; h < kH; ++h) s += h_last[(size_t)b * kH + h] * W[(size_t)c * kH + h];
  out[idx] = s;
}

__global__ void fill_f32(float* p, float v, int n) {
  int i = blockIdx.x * blockDim.x + threadIdx.x;
  if (i < n) p[i] = v;
}
__global__ void fill_f16(_Float16* p, int n) {
  int i = blockIdx.x * blockDim.x + threadIdx.x;
  if (i < n) p[i] = (_Float16)0.0f;
}

// ---------------------------------------------------------------------------
extern "C" void kernel_launch(void* const* d_in, const int* in_sizes, int n_in,
                              void* d_out, int out_size, void* d_ws, size_t ws_size,
                              hipStream_t stream) {
  (void)in_sizes; (void)n_in; (void)out_size; (void)ws_size;
  const int* premise = (const int*)d_in[0];
  const int* p_len = (const int*)d_in[1];
  const int* hyp = (const int*)d_in[2];
  const int* h_len = (const int*)d_in[3];
  const float* embed = (const float*)d_in[4];
  const float* p_Wih = (const float*)d_in[5];
  const float* p_Whh = (const float*)d_in[6];
  const float* p_b = (const float*)d_in[7];
  const float* h_Wih = (const float*)d_in[8];
  const float* h_Whh = (const float*)d_in[9];
  const float* h_b = (const float*)d_in[10];
  const float* m_Wih = (const float*)d_in[11];
  const float* m_Whh = (const float*)d_in[12];
  const float* m_b = (const float*)d_in[13];
  const float* w_e = (const float*)d_in[14];
  const float* Ws = (const float*)d_in[15];
  const float* Wt = (const float*)d_in[16];
  const float* Wm = (const float*)d_in[17];
  const float* fc_W = (const float*)d_in[18];
  const float* fc_b = (const float*)d_in[19];
  float* out = (float*)d_out;

  char* base = (char*)d_ws;
  size_t off = 0;
  auto alloc = [&](size_t bytes) -> char* {
    char* p = base + off;
    off = (off + bytes + 255) & ~(size_t)255;
    return p;
  };

  _Float16* Xp = (_Float16*)alloc((size_t)kSP * STEP_FRAG_X * 2);   // premise x frags
  _Float16* Xh = (_Float16*)alloc((size_t)kSH * STEP_FRAG_X * 2);   // hypothesis x frags
  _Float16* Bp = (_Float16*)alloc((size_t)128 * 26 * FRAG * 2);     // [Wih_p^T;Whh_p^T]
  _Float16* Bh = (_Float16*)alloc((size_t)128 * 26 * FRAG * 2);     // [Wih_h^T;Whh_h^T]
  _Float16* Bq = (_Float16*)alloc((size_t)32 * 32 * FRAG * 2);      // [Wt^T;Wm^T]
  _Float16* Bm = (_Float16*)alloc((size_t)128 * 48 * FRAG * 2);     // [m_Wih^T;m_Whh^T]
  _Float16* BWs = (_Float16*)alloc((size_t)32 * 16 * FRAG * 2);     // Ws^T
  float* hs_f32 = (float*)alloc((size_t)kSP * kB * kH * 4);         // masked premise h_s
  _Float16* hs_frag = (_Float16*)alloc((size_t)kSP * STEP_FRAG_H * 2);
  _Float16* ht_frag = (_Float16*)alloc((size_t)kSH * STEP_FRAG_H * 2);
  float* WsHs = (float*)alloc((size_t)kSP * kB * kH * 4);
  float* gates = (float*)alloc((size_t)kB * kG * 4);
  float* c_state = (float*)alloc((size_t)kB * kH * 4);
  _Float16* hcarry = (_Float16*)alloc((size_t)STEP_FRAG_H * 2);
  float* qbuf = (float*)alloc((size_t)kB * kH * 4);
  float* e_buf = (float*)alloc((size_t)kSP * kB * 4);
  float* alpha = (float*)alloc((size_t)kSP * kB * 4);
  _Float16* a_frag = (_Float16*)alloc((size_t)STEP_FRAG_H * 2);
  float* h_last = (float*)alloc((size_t)kB * kH * 4);

  auto nb = [](int n) { return (n + 255) / 256; };

  // --- Pack weights into B-operand fragment buffers (combined-K layouts) ----
  pack_b_seg<<<nb(128 * 10 * 32), 256, 0, stream>>>(p_Wih, kG, kE, Bp, 0, 26, 10, 128);
  pack_b_seg<<<nb(128 * 16 * 32), 256, 0, stream>>>(p_Whh, kG, kH, Bp, 10, 26, 16, 128);
  pack_b_seg<<<nb(128 * 10 * 32), 256, 0, stream>>>(h_Wih, kG, kE, Bh, 0, 26, 10, 128);
  pack_b_seg<<<nb(128 * 16 * 32), 256, 0, stream>>>(h_Whh, kG, kH, Bh, 10, 26, 16, 128);
  pack_b_seg<<<nb(32 * 16 * 32), 256, 0, stream>>>(Wt, kH, kH, Bq, 0, 32, 16, 32);
  pack_b_seg<<<nb(32 * 16 * 32), 256, 0, stream>>>(Wm, kH, kH, Bq, 16, 32, 16, 32);
  pack_b_seg<<<nb(128 * 32 * 32), 256, 0, stream>>>(m_Wih, kG, 2 * kH, Bm, 0, 48, 32, 128);
  pack_b_seg<<<nb(128 * 16 * 32), 256, 0, stream>>>(m_Whh, kG, kH, Bm, 32, 48, 16, 128);
  pack_b_seg<<<nb(32 * 16 * 32), 256, 0, stream>>>(Ws, kH, kH, BWs, 0, 16, 16, 32);

  // --- Embedding gather into A fragments ------------------------------------
  gather_embed<<<nb((kSP * kB / 16) * KB_E * 32), 256, 0, stream>>>(premise, kSP * kB, embed, Xp);
  gather_embed<<<nb((kSH * kB / 16) * KB_E * 32), 256, 0, stream>>>(hyp, kSH * kB, embed, Xh);

  const int nBH = kB * kH;  // 65536

  // --- Premise encoder LSTM (64 sequential steps) ---------------------------
  fill_f32<<<nb(nBH), 256, 0, stream>>>(c_state, 0.0f, nBH);
  fill_f16<<<nb(nBH), 256, 0, stream>>>(hcarry, nBH);
  for (int t = 0; t < kSP; ++t) {
    wmma_gemm<KB_E, KB_H, 0><<<32, 256, 0, stream>>>(Xp + (size_t)t * STEP_FRAG_X, hcarry,
                                                     nullptr, Bp, p_b, gates, 8, 128, kG);
    lstm_pointwise<<<nb(nBH), 256, 0, stream>>>(gates, c_state, hcarry, t, p_len,
                                                hs_f32 + (size_t)t * nBH,
                                                hs_frag + (size_t)t * STEP_FRAG_H);
  }

  // --- Precompute Ws_hs = h_s @ Ws^T  (one big M=8192 GEMM) -----------------
  wmma_gemm<KB_H, 0, 0><<<512, 256, 0, stream>>>(hs_frag, nullptr, nullptr, BWs, nullptr,
                                                 WsHs, kSP * 8, 32, kH);

  // --- Hypothesis encoder LSTM (32 sequential steps) ------------------------
  fill_f32<<<nb(nBH), 256, 0, stream>>>(c_state, 0.0f, nBH);
  fill_f16<<<nb(nBH), 256, 0, stream>>>(hcarry, nBH);
  for (int t = 0; t < kSH; ++t) {
    wmma_gemm<KB_E, KB_H, 0><<<32, 256, 0, stream>>>(Xh + (size_t)t * STEP_FRAG_X, hcarry,
                                                     nullptr, Bh, h_b, gates, 8, 128, kG);
    lstm_pointwise<<<nb(nBH), 256, 0, stream>>>(gates, c_state, hcarry, t, h_len,
                                                nullptr, ht_frag + (size_t)t * STEP_FRAG_H);
  }

  // --- Match-LSTM with attention (32 sequential steps) ----------------------
  fill_f32<<<nb(nBH), 256, 0, stream>>>(c_state, 0.0f, nBH);
  fill_f16<<<nb(nBH), 256, 0, stream>>>(hcarry, nBH);  // h_m frags
  fill_f32<<<nb(nBH), 256, 0, stream>>>(h_last, 0.0f, nBH);
  for (int k = 0; k < kSH; ++k) {
    // q = h_t_k @ Wt^T + h_m @ Wm^T
    wmma_gemm<KB_H, KB_H, 0><<<8, 256, 0, stream>>>(ht_frag + (size_t)k * STEP_FRAG_H, hcarry,
                                                    nullptr, Bq, nullptr, qbuf, 8, 32, kH);
    attn_e<<<nb(kSP * kB * 32), 256, 0, stream>>>(WsHs, qbuf, w_e, e_buf);
    attn_softmax<<<nb(kB * 32), 256, 0, stream>>>(e_buf, alpha);
    attn_ak<<<nb(nBH), 256, 0, stream>>>(alpha, hs_f32, a_frag);
    // gates = [a_k | h_t_k] @ m_Wih^T + h_m @ m_Whh^T + m_b
    wmma_gemm<KB_H, KB_H, KB_H><<<32, 256, 0, stream>>>(a_frag,
                                                        ht_frag + (size_t)k * STEP_FRAG_H,
                                                        hcarry, Bm, m_b, gates, 8, 128, kG);
    match_pointwise<<<nb(nBH), 256, 0, stream>>>(gates, c_state, hcarry, h_last, k, h_len);
  }

  // --- Final FC -------------------------------------------------------------
  fc_out<<<nb(kB * 3), 256, 0, stream>>>(h_last, fc_W, fc_b, out);
}
